// TRMPolicyValue_28106265985045
// MI455X (gfx1250) — compile-verified, matrix operand order checked by reading, not run on hardware
//
#include <hip/hip_runtime.h>
#include <hip/hip_bf16.h>
#include <math.h>

typedef _Float16 v16h __attribute__((ext_vector_type(16)));
typedef _Float16 v8h  __attribute__((ext_vector_type(8)));
typedef _Float16 h2   __attribute__((ext_vector_type(2)));
typedef float    v8f  __attribute__((ext_vector_type(8)));

#define DD 512
#define B_ 128
#define NV_ 1024
#define NC_ 64
#define NA_ 1024
#define NFAM_ 8
#define NVTOK (B_ * NV_)   /* 131072 */
#define NATOK (B_ * NA_)   /* 131072 */
#define NCTOK (B_ * NC_)   /* 8192   */

#define BM 128
#define BN 64
#define BK 32
#define FMAX 28
#define ASTR 40   /* padded LDS row stride in halves (80 B: 16B-aligned, bank-spread) */

__device__ __forceinline__ float gelu_exact(float x) {
  return 0.5f * x * (1.0f + erff(x * 0.70710678118654752f));
}
__device__ __forceinline__ float sigmoidf_(float x) {
  return 1.0f / (1.0f + expf(-x));
}

// permute K-group g (8 halves) to fragment position group: 0->0, 1->2, 2->1, 3->3
__device__ __forceinline__ int pgrp(int g) { return ((g & 1) << 1) | (g >> 1); }

// one v16h fragment = two 16B LDS reads (ds_load_b128) + shuffle-combine
__device__ __forceinline__ v16h load_frag16(const _Float16* p) {
  v8h lo = *(const v8h*)p;
  v8h hi = *(const v8h*)(p + 8);
  return __builtin_shufflevector(lo, hi, 0, 1, 2, 3, 4, 5, 6, 7,
                                 8, 9, 10, 11, 12, 13, 14, 15);
}

// async global->LDS 16-byte chunk copy (per-lane addresses), ASYNCcnt-tracked
__device__ __forceinline__ void async_copy16(const _Float16* gsrc, _Float16* ldst) {
  unsigned laddr = (unsigned)(size_t)ldst;
  asm volatile("global_load_async_to_lds_b128 %0, %1, off"
               :: "v"(laddr), "v"(gsrc) : "memory");
}
__device__ __forceinline__ void wait_async0() {
  asm volatile("s_wait_asynccnt 0x0" ::: "memory");
}

// ---------------------------------------------------------------------------
// Fused token-projection MLP2:  Y = gelu(X @ W1 + b1) @ W2 + b2
// Hidden tile computed in-register -> permuted f16 LDS; 8 waves = 4(M)x2(N),
// each wave 2x2 WMMA 16x16x32 tiles. Optional f16 mirror output (Y16),
// stored in a single uniformly-branched epilogue loop.
// ---------------------------------------------------------------------------
__global__ __launch_bounds__(256)
void proj_mlp2_kernel(const float* __restrict__ X, int F,
                      const float* __restrict__ W1, const float* __restrict__ b1,
                      const float* __restrict__ W2, const float* __restrict__ b2,
                      float* __restrict__ Y, _Float16* __restrict__ Y16, int M)
{
  __shared__ float    sX[BM * FMAX];
  __shared__ float    sW1[FMAX * DD];
  __shared__ _Float16 sA[BM * ASTR] __attribute__((aligned(16)));
  __shared__ _Float16 sB[BN * ASTR] __attribute__((aligned(16)));

  const int tid = threadIdx.x;
  const int m0 = blockIdx.x * BM;
  const int n0 = blockIdx.y * BN;

  for (int i = tid; i < BM * F; i += 256) {
    int r = i / F, f = i % F;
    sX[r * F + f] = X[(size_t)(m0 + r) * F + f];
  }
  for (int i = tid; i < F * DD; i += 256) sW1[i] = W1[i];
  __syncthreads();

  const int w    = tid >> 5;
  const int lane = tid & 31;
  const int wm   = w & 3;
  const int wn   = w >> 2;
  const int l15  = lane & 15;
  const int foff = (lane < 16) ? 0 : 16;   // fragment half-select (ISA K split)

  v8f zero = {0.f, 0.f, 0.f, 0.f, 0.f, 0.f, 0.f, 0.f};
  v8f acc[2][2];
  acc[0][0] = zero; acc[0][1] = zero; acc[1][0] = zero; acc[1][1] = zero;

  for (int k0 = 0; k0 < DD; k0 += BK) {
    __syncthreads();
    // hidden tile 128x32, two values per store (h2 -> ds_store_b32)
    for (int e = tid; e < BM * (BK / 2); e += 256) {
      int r = e >> 4, cc = (e & 15) * 2;
      float s0 = b1[k0 + cc], s1 = b1[k0 + cc + 1];
      const float* xr = &sX[r * F];
      for (int f = 0; f < F; ++f) {
        float xv = xr[f];
        s0 += xv * sW1[f * DD + (k0 + cc)];
        s1 += xv * sW1[f * DD + (k0 + cc + 1)];
      }
      h2 hv; hv[0] = (_Float16)gelu_exact(s0); hv[1] = (_Float16)gelu_exact(s1);
      int pos = pgrp(cc >> 3) * 8 + (cc & 7);
      *(h2*)&sA[r * ASTR + pos] = hv;
    }
    // W2 tile -> permuted f16 LDS (coalesced global reads)
    for (int i = tid; i < BK * BN; i += 256) {
      int kk = i / BN, nn = i % BN;
      int pos = pgrp(kk >> 3) * 8 + (kk & 7);
      sB[nn * ASTR + pos] = (_Float16)W2[(size_t)(k0 + kk) * DD + (n0 + nn)];
    }
    __syncthreads();

    #pragma unroll
    for (int tm = 0; tm < 2; ++tm) {
      v16h af = load_frag16(&sA[(wm * 32 + tm * 16 + l15) * ASTR + foff]);
      #pragma unroll
      for (int tn = 0; tn < 2; ++tn) {
        v16h bf = load_frag16(&sB[(wn * 32 + tn * 16 + l15) * ASTR + foff]);
        acc[tm][tn] = __builtin_amdgcn_wmma_f32_16x16x32_f16(
            false, af, false, bf, (short)0, acc[tm][tn], false, false);
      }
    }
  }

  // epilogue: biased results into a register array, f32 stores, then one
  // uniform branch for the optional f16 mirror stores.
  float res[2][2][8];
  #pragma unroll
  for (int tm = 0; tm < 2; ++tm)
    #pragma unroll
    for (int tn = 0; tn < 2; ++tn) {
      int nc = n0 + wn * 32 + tn * 16 + l15;
      #pragma unroll
      for (int r = 0; r < 8; ++r) res[tm][tn][r] = acc[tm][tn][r] + b2[nc];
    }

  #pragma unroll
  for (int tm = 0; tm < 2; ++tm)
    #pragma unroll
    for (int tn = 0; tn < 2; ++tn) {
      int mbase = m0 + wm * 32 + tm * 16 + ((lane < 16) ? 0 : 8);
      int nc    = n0 + wn * 32 + tn * 16 + l15;
      #pragma unroll
      for (int r = 0; r < 8; ++r)
        Y[(size_t)(mbase + r) * DD + nc] = res[tm][tn][r];
    }

  if (Y16) {
    #pragma unroll
    for (int tm = 0; tm < 2; ++tm)
      #pragma unroll
      for (int tn = 0; tn < 2; ++tn) {
        int mbase = m0 + wm * 32 + tm * 16 + ((lane < 16) ? 0 : 8);
        int nc    = n0 + wn * 32 + tn * 16 + l15;
        #pragma unroll
        for (int r = 0; r < 8; ++r)
          Y16[(size_t)(mbase + r) * DD + nc] = (_Float16)res[tm][tn][r];
      }
  }
}

// ---------------------------------------------------------------------------
// WMMA GEMM, f16 operands: Y[M,N] = A[M,K] @ Bt[N,K]^T.
// Global->LDS staging via GLOBAL_LOAD_ASYNC_TO_LDS_B128 (permutation is
// 16B-chunk granular, so the async DMA lands data fragment-ready).
// ---------------------------------------------------------------------------
__global__ __launch_bounds__(256)
void wmma_gemm_f16_kernel(const _Float16* __restrict__ A, const _Float16* __restrict__ Bt,
                          float* __restrict__ Y, int M, int N, int K)
{
  __shared__ _Float16 sA[BM * ASTR] __attribute__((aligned(16)));
  __shared__ _Float16 sB[BN * ASTR] __attribute__((aligned(16)));

  const int tid = threadIdx.x;
  const int m0 = blockIdx.x * BM;
  const int n0 = blockIdx.y * BN;

  const int w    = tid >> 5;
  const int lane = tid & 31;
  const int wm   = w & 3;
  const int wn   = w >> 2;
  const int l15  = lane & 15;
  const int foff = (lane < 16) ? 0 : 16;

  v8f zero = {0.f, 0.f, 0.f, 0.f, 0.f, 0.f, 0.f, 0.f};
  v8f acc[2][2];
  acc[0][0] = zero; acc[0][1] = zero; acc[1][0] = zero; acc[1][1] = zero;

  for (int k0 = 0; k0 < K; k0 += BK) {
    __syncthreads();
    // A: 128 rows x 4 chunks = 512 chunks (2/thread); B: 64 x 4 = 256 (1/thread)
    for (int t = tid; t < BM * 4; t += 256) {
      int r = t >> 2, gg = t & 3;
      async_copy16(&A[(size_t)(m0 + r) * K + k0 + gg * 8], &sA[r * ASTR + pgrp(gg) * 8]);
    }
    for (int t = tid; t < BN * 4; t += 256) {
      int r = t >> 2, gg = t & 3;
      async_copy16(&Bt[(size_t)(n0 + r) * K + k0 + gg * 8], &sB[r * ASTR + pgrp(gg) * 8]);
    }
    wait_async0();
    __syncthreads();

    #pragma unroll
    for (int tm = 0; tm < 2; ++tm) {
      v16h af = load_frag16(&sA[(wm * 32 + tm * 16 + l15) * ASTR + foff]);
      #pragma unroll
      for (int tn = 0; tn < 2; ++tn) {
        v16h bf = load_frag16(&sB[(wn * 32 + tn * 16 + l15) * ASTR + foff]);
        acc[tm][tn] = __builtin_amdgcn_wmma_f32_16x16x32_f16(
            false, af, false, bf, (short)0, acc[tm][tn], false, false);
      }
    }
  }

  #pragma unroll
  for (int tm = 0; tm < 2; ++tm)
    #pragma unroll
    for (int tn = 0; tn < 2; ++tn) {
      int mbase = m0 + wm * 32 + tm * 16 + ((lane < 16) ? 0 : 8);
      int nc    = n0 + wn * 32 + tn * 16 + l15;
      #pragma unroll
      for (int r = 0; r < 8; ++r)
        Y[(size_t)(mbase + r) * N + nc] = acc[tm][tn][r];
    }
}

// pol_W1[D:,:] ([K,N] f32) -> Bt[N,K] f16
__global__ void build_bt16_kernel(const float* __restrict__ W, _Float16* __restrict__ Bt)
{
  int idx = blockIdx.x * blockDim.x + threadIdx.x;
  if (idx >= DD * DD) return;
  int n = idx / DD, k = idx % DD;
  Bt[(size_t)n * DD + k] = (_Float16)W[(size_t)k * DD + n];
}

// ---------------------------------------------------------------------------
// Small dense kernels (B=128 rows: GRU, heads).
// ---------------------------------------------------------------------------
__global__ void matmul_nn_small(const float* __restrict__ X, const float* __restrict__ W,
                                const float* __restrict__ bias, float* __restrict__ Y,
                                int M, int K, int N, int act)
{
  int n = blockIdx.y * blockDim.x + threadIdx.x;
  int m = blockIdx.x;
  if (n >= N) return;
  const float* xr = X + (size_t)m * K;
  float s = bias ? bias[n] : 0.0f;
  for (int k = 0; k < K; ++k) s += xr[k] * W[(size_t)k * N + n];
  if (act) s = gelu_exact(s);
  Y[(size_t)m * N + n] = s;
}

__global__ void matmul_nt_small(const float* __restrict__ X, const float* __restrict__ W,
                                const float* __restrict__ bias, float* __restrict__ Y,
                                int M, int K, int N)
{
  int n = blockIdx.y * blockDim.x + threadIdx.x;
  int m = blockIdx.x;
  if (n >= N) return;
  const float* xr = X + (size_t)m * K;
  const float* wr = W + (size_t)n * K;
  float s = bias ? bias[n] : 0.0f;
  for (int k = 0; k < K; ++k) s += xr[k] * wr[k];
  Y[(size_t)m * N + n] = s;
}

__global__ void gru_combine_kernel(const float* __restrict__ gi, const float* __restrict__ gh,
                                   const float* __restrict__ h, float* __restrict__ out)
{
  int idx = blockIdx.x * blockDim.x + threadIdx.x;
  if (idx >= B_ * DD) return;
  int b = idx / DD, k = idx % DD;
  const float* gib = gi + (size_t)b * 3 * DD;
  const float* ghb = gh + (size_t)b * 3 * DD;
  float r  = sigmoidf_(gib[k] + ghb[k]);
  float zg = sigmoidf_(gib[DD + k] + ghb[DD + k]);
  float nn = tanhf(gib[2 * DD + k] + r * ghb[2 * DD + k]);
  out[idx] = (1.0f - zg) * nn + zg * h[idx];
}

__global__ __launch_bounds__(256)
void layernorm_kernel(const float* __restrict__ X, const float* __restrict__ gw,
                      const float* __restrict__ bw, float* __restrict__ Y)
{
  __shared__ float red[256];
  __shared__ float s_mean, s_rstd;
  int row = blockIdx.x, tid = threadIdx.x;
  float v0 = X[(size_t)row * DD + tid];
  float v1 = X[(size_t)row * DD + 256 + tid];
  red[tid] = v0 + v1;
  __syncthreads();
  for (int s = 128; s > 0; s >>= 1) {
    if (tid < s) red[tid] += red[tid + s];
    __syncthreads();
  }
  if (tid == 0) s_mean = red[0] / (float)DD;
  __syncthreads();
  float m = s_mean;
  float d0 = v0 - m, d1 = v1 - m;
  red[tid] = d0 * d0 + d1 * d1;
  __syncthreads();
  for (int s = 128; s > 0; s >>= 1) {
    if (tid < s) red[tid] += red[tid + s];
    __syncthreads();
  }
  if (tid == 0) s_rstd = rsqrtf(red[0] / (float)DD + 1e-5f);
  __syncthreads();
  float rs = s_rstd;
  Y[(size_t)row * DD + tid]       = d0 * rs * gw[tid] + bw[tid];
  Y[(size_t)row * DD + 256 + tid] = d1 * rs * gw[256 + tid] + bw[256 + tid];
}

// ---------------------------------------------------------------------------
// Gated pool: attn = masked_softmax(x @ W + b); out = attn^T x.  One block/batch.
// ---------------------------------------------------------------------------
__global__ __launch_bounds__(256)
void gated_pool_kernel(const float* __restrict__ X, const unsigned char* __restrict__ mask,
                       const float* __restrict__ W, const float* __restrict__ bsc,
                       float* __restrict__ out, int N)
{
  __shared__ float sc[1024];
  __shared__ float red[256];
  __shared__ float s_bcast;
  int b = blockIdx.x, tid = threadIdx.x;
  const float* xb = X + (size_t)b * N * DD;
  float bias = bsc[0];

  for (int n = tid; n < N; n += 256) {
    float s = bias;
    const float* xr = xb + (size_t)n * DD;
    for (int k = 0; k < DD; ++k) s += xr[k] * W[k];
    sc[n] = mask[(size_t)b * N + n] ? s : -3.402823466e38f;
  }
  __syncthreads();

  float mx = -3.402823466e38f;
  for (int n = tid; n < N; n += 256) mx = fmaxf(mx, sc[n]);
  red[tid] = mx;
  __syncthreads();
  for (int s = 128; s > 0; s >>= 1) {
    if (tid < s) red[tid] = fmaxf(red[tid], red[tid + s]);
    __syncthreads();
  }
  if (tid == 0) s_bcast = red[0];
  __syncthreads();
  float gmax = s_bcast;
  __syncthreads();

  float ssum = 0.0f;
  for (int n = tid; n < N; n += 256) {
    float e = mask[(size_t)b * N + n] ? expf(sc[n] - gmax) : 0.0f;
    sc[n] = e;
    ssum += e;
  }
  red[tid] = ssum;
  __syncthreads();
  for (int s = 128; s > 0; s >>= 1) {
    if (tid < s) red[tid] += red[tid + s];
    __syncthreads();
  }
  if (tid == 0) s_bcast = 1.0f / fmaxf(red[0], 1e-9f);
  __syncthreads();
  float inv = s_bcast;

  for (int d = tid; d < DD; d += 256) {
    float acc = 0.0f;
    for (int n = 0; n < N; ++n) acc += sc[n] * xb[(size_t)n * DD + d];
    out[(size_t)b * DD + d] = acc * inv;
  }
}

// ---------------------------------------------------------------------------
// Policy logits: one token per wave32; wave-reduce over D=512.
// ---------------------------------------------------------------------------
__global__ __launch_bounds__(256)
void policy_logits_kernel(const float* __restrict__ apart, const float* __restrict__ zpart,
                          const float* __restrict__ pw2, const float* __restrict__ pb2,
                          const unsigned char* __restrict__ amask, float* __restrict__ logits)
{
  int w = threadIdx.x >> 5, lane = threadIdx.x & 31;
  int token = blockIdx.x * 8 + w;
  int b = token >> 10;
  const float* ap = apart + (size_t)token * DD;
  const float* zp = zpart + (size_t)b * DD;
  float acc = 0.0f;
  #pragma unroll
  for (int i = 0; i < 16; ++i) {
    int k = lane + 32 * i;
    float h = gelu_exact(ap[k] + zp[k]);
    acc += h * pw2[k];
  }
  #pragma unroll
  for (int off = 16; off > 0; off >>= 1) acc += __shfl_xor(acc, off, 32);
  if (lane == 0) {
    float lg = acc + pb2[0];
    logits[token] = amask[token] ? lg : -3.402823466e38f;
  }
}

// ---------------------------------------------------------------------------
// Masked softmax over NA + pooled y[b,:] = probs^T a[b].  One block/batch.
// ---------------------------------------------------------------------------
__global__ __launch_bounds__(256)
void softmax_y_kernel(const float* __restrict__ logits, const unsigned char* __restrict__ amask,
                      const float* __restrict__ a, float* __restrict__ y)
{
  __shared__ float sc[NA_];
  __shared__ float red[256];
  __shared__ float s_bcast;
  int b = blockIdx.x, tid = threadIdx.x;

  float mx = -3.402823466e38f;
  for (int n = tid; n < NA_; n += 256) {
    float l = logits[(size_t)b * NA_ + n];
    sc[n] = l;
    mx = fmaxf(mx, l);
  }
  red[tid] = mx;
  __syncthreads();
  for (int s = 128; s > 0; s >>= 1) {
    if (tid < s) red[tid] = fmaxf(red[tid], red[tid + s]);
    __syncthreads();
  }
  if (tid == 0) s_bcast = red[0];
  __syncthreads();
  float gmax = s_bcast;
  __syncthreads();

  float ssum = 0.0f;
  for (int n = tid; n < NA_; n += 256) {
    float e = amask[(size_t)b * NA_ + n] ? expf(sc[n] - gmax) : 0.0f;
    sc[n] = e;
    ssum += e;
  }
  red[tid] = ssum;
  __syncthreads();
  for (int s = 128; s > 0; s >>= 1) {
    if (tid < s) red[tid] += red[tid + s];
    __syncthreads();
  }
  if (tid == 0) s_bcast = 1.0f / fmaxf(red[0], 1e-9f);
  __syncthreads();
  float inv = s_bcast;

  const float* ab = a + (size_t)b * NA_ * DD;
  for (int d = tid; d < DD; d += 256) {
    float acc = 0.0f;
    for (int n = 0; n < NA_; ++n) acc += sc[n] * ab[(size_t)n * DD + d];
    y[(size_t)b * DD + d] = acc * inv;
  }
}

__global__ void concat4_kernel(const float* __restrict__ p0, const float* __restrict__ p1,
                               const float* __restrict__ p2, const float* __restrict__ p3,
                               float* __restrict__ out)
{
  int idx = blockIdx.x * blockDim.x + threadIdx.x;
  if (idx >= B_ * 4 * DD) return;
  int b = idx / (4 * DD), j = idx % (4 * DD);
  float v;
  if (j < DD)            v = p0[(size_t)b * DD + j];
  else if (j < 2 * DD)   v = p1[(size_t)b * DD + (j - DD)];
  else if (j < 3 * DD)   v = p2[(size_t)b * DD + (j - 2 * DD)];
  else                   v = p3[(size_t)b * DD + (j - 3 * DD)];
  out[idx] = v;
}

__global__ void copy_kernel(float* __restrict__ dst, const float* __restrict__ src, int n)
{
  int idx = blockIdx.x * blockDim.x + threadIdx.x;
  if (idx < n) dst[idx] = src[idx];
}

// ---------------------------------------------------------------------------
extern "C" void kernel_launch(void* const* d_in, const int* in_sizes, int n_in,
                              void* d_out, int out_size, void* d_ws, size_t ws_size,
                              hipStream_t stream)
{
  const float* gfeat = (const float*)d_in[0];
  const float* vtok  = (const float*)d_in[1];
  const float* ctok  = (const float*)d_in[2];
  const float* atok  = (const float*)d_in[3];
  const unsigned char* vmask = (const unsigned char*)d_in[4];
  const unsigned char* cmask = (const unsigned char*)d_in[5];
  const unsigned char* amask = (const unsigned char*)d_in[6];
#define P(i) ((const float*)d_in[i])
  const float *gp_W1 = P(7),  *gp_b1 = P(8),  *gp_W2 = P(9),  *gp_b2 = P(10);
  const float *vp_W1 = P(11), *vp_b1 = P(12), *vp_W2 = P(13), *vp_b2 = P(14);
  const float *cp_W1 = P(15), *cp_b1 = P(16), *cp_W2 = P(17), *cp_b2 = P(18);
  const float *ap_W1 = P(19), *ap_b1 = P(20), *ap_W2 = P(21), *ap_b2 = P(22);
  const float *vpool_W = P(23), *vpool_b = P(24);
  const float *cpool_W = P(25), *cpool_b = P(26);
  const float *apool_W = P(27), *apool_b = P(28);
  const float *init_W1 = P(29), *init_b1 = P(30), *init_W2 = P(31), *init_b2 = P(32);
  const float *Wih = P(33), *Whh = P(34), *bih = P(35), *bhh = P(36);
  const float *pol_W1 = P(37), *pol_b1 = P(38), *pol_W2 = P(39), *pol_b2 = P(40);
  const float *val_W1 = P(41), *val_b1 = P(42), *val_W2 = P(43), *val_b2 = P(44);
  const float *fam_W1 = P(45), *fam_b1 = P(46), *fam_W2 = P(47), *fam_b2 = P(48);
  const float *ln_g = P(49), *ln_b = P(50);
#undef P

  float* ws = (float*)d_ws;
  size_t o = 0;
  float* v     = ws + o; o += (size_t)NVTOK * DD;
  float* a     = ws + o; o += (size_t)NATOK * DD;
  float* apart = ws + o; o += (size_t)NATOK * DD;
  float* c     = ws + o; o += (size_t)NCTOK * DD;
  float* g     = ws + o; o += (size_t)B_ * DD;
  float* vpool = ws + o; o += (size_t)B_ * DD;
  float* cpool = ws + o; o += (size_t)B_ * DD;
  float* apool = ws + o; o += (size_t)B_ * DD;
  float* ybuf  = ws + o; o += (size_t)B_ * DD;
  float* z     = ws + o; o += (size_t)B_ * DD;
  float* znew  = ws + o; o += (size_t)B_ * DD;
  float* zpart = ws + o; o += (size_t)B_ * DD;
  float* hid   = ws + o; o += (size_t)B_ * DD;
  float* inp   = ws + o; o += (size_t)B_ * 4 * DD;
  float* gi    = ws + o; o += (size_t)B_ * 3 * DD;
  float* gh    = ws + o; o += (size_t)B_ * 3 * DD;
  _Float16* a16  = (_Float16*)(ws + o); o += (size_t)NATOK * DD / 2;  // f16 mirror of a
  _Float16* bt16 = (_Float16*)(ws + o); o += (size_t)DD * DD / 2;     // pol_W1[D:]^T f16

  float* out      = (float*)d_out;
  float* logits   = out;                       // [128, 1024]
  float* value    = out + (size_t)B_ * NA_;    // [128]
  float* fam      = value + B_;                // [128, 8]

  // --- token projections (fused MLP2, WMMA) ---
  proj_mlp2_kernel<<<dim3(NVTOK / BM, DD / BN), 256, 0, stream>>>(
      vtok, 16, vp_W1, vp_b1, vp_W2, vp_b2, v, (_Float16*)nullptr, NVTOK);
  proj_mlp2_kernel<<<dim3(NATOK / BM, DD / BN), 256, 0, stream>>>(
      atok, 28, ap_W1, ap_b1, ap_W2, ap_b2, a, a16, NATOK);
  proj_mlp2_kernel<<<dim3(NCTOK / BM, DD / BN), 256, 0, stream>>>(
      ctok, 12, cp_W1, cp_b1, cp_W2, cp_b2, c, (_Float16*)nullptr, NCTOK);
  proj_mlp2_kernel<<<dim3(B_ / BM, DD / BN), 256, 0, stream>>>(
      gfeat, 24, gp_W1, gp_b1, gp_W2, gp_b2, g, (_Float16*)nullptr, B_);

  // --- gated pools ---
  gated_pool_kernel<<<B_, 256, 0, stream>>>(v, vmask, vpool_W, vpool_b, vpool, NV_);
  gated_pool_kernel<<<B_, 256, 0, stream>>>(c, cmask, cpool_W, cpool_b, cpool, NC_);
  gated_pool_kernel<<<B_, 256, 0, stream>>>(a, amask, apool_W, apool_b, apool, NA_);

  // --- init state z = MLP2(concat(g, vpool, cpool, apool)) ---
  concat4_kernel<<<(B_ * 4 * DD + 255) / 256, 256, 0, stream>>>(g, vpool, cpool, apool, inp);
  matmul_nn_small<<<dim3(B_, (DD + 255) / 256), 256, 0, stream>>>(
      inp, init_W1, init_b1, hid, B_, 4 * DD, DD, 1);
  matmul_nn_small<<<dim3(B_, (DD + 255) / 256), 256, 0, stream>>>(
      hid, init_W2, init_b2, z, B_, DD, DD, 0);

  // --- hoisted loop-invariant policy term: apart = a @ pol_W1[D:] (async + WMMA) ---
  build_bt16_kernel<<<(DD * DD + 255) / 256, 256, 0, stream>>>(pol_W1 + (size_t)DD * DD, bt16);
  wmma_gemm_f16_kernel<<<dim3(NATOK / BM, DD / BN), 256, 0, stream>>>(
      a16, bt16, apart, NATOK, DD, DD);

  // y0 = apool
  copy_kernel<<<(B_ * DD + 255) / 256, 256, 0, stream>>>(ybuf, apool, B_ * DD);

  // --- refine loop ---
  for (int step = 0; step < 3; ++step) {
    concat4_kernel<<<(B_ * 4 * DD + 255) / 256, 256, 0, stream>>>(g, vpool, cpool, ybuf, inp);
    matmul_nt_small<<<dim3(B_, (3 * DD + 255) / 256), 256, 0, stream>>>(
        inp, Wih, bih, gi, B_, 4 * DD, 3 * DD);
    matmul_nt_small<<<dim3(B_, (3 * DD + 255) / 256), 256, 0, stream>>>(
        z, Whh, bhh, gh, B_, DD, 3 * DD);
    gru_combine_kernel<<<(B_ * DD + 255) / 256, 256, 0, stream>>>(gi, gh, z, znew);
    layernorm_kernel<<<B_, 256, 0, stream>>>(znew, ln_g, ln_b, z);
    matmul_nn_small<<<dim3(B_, (DD + 255) / 256), 256, 0, stream>>>(
        z, pol_W1, pol_b1, zpart, B_, DD, DD, 0);
    policy_logits_kernel<<<NATOK / 8, 256, 0, stream>>>(
        apart, zpart, pol_W2, pol_b2, amask, logits);
    softmax_y_kernel<<<B_, 256, 0, stream>>>(logits, amask, a, ybuf);
    matmul_nn_small<<<dim3(B_, (DD + 255) / 256), 256, 0, stream>>>(
        z, val_W1, val_b1, hid, B_, DD, DD, 1);
    matmul_nn_small<<<dim3(B_, 1), 256, 0, stream>>>(
        hid, val_W2, val_b2, value, B_, DD, 1, 0);
  }

  // --- family head ---
  matmul_nn_small<<<dim3(B_, (DD + 255) / 256), 256, 0, stream>>>(
      z, fam_W1, fam_b1, hid, B_, DD, DD, 1);
  matmul_nn_small<<<dim3(B_, 1), 256, 0, stream>>>(
      hid, fam_W2, fam_b2, fam, B_, DD, NFAM_, 0);
}
